// RNNModel_1254130450618
// MI455X (gfx1250) — compile-verified, matrix operand order checked by reading, not run on hardware
//
#include <hip/hip_runtime.h>
#include <math.h>

#define VOCAB  128
#define EMBED  128
#define HIDDEN 256
#define BATCH  512
#define SEQ    512
#define NSTEP  (SEQ - 1)

// LDS row strides (padded to spread rows across the 64 banks)
#define HSTR 264   // u16 stride for h / P / Why rows   (264*2B = 528B rows)
#define XSTR 520   // int stride for x rows
#define BSTR 264   // f32 stride for baseline rows

typedef __attribute__((ext_vector_type(16))) __bf16 v16bf;
typedef __attribute__((ext_vector_type(8)))  float  v8f;

union Frag {
    v16bf v;
    uint4 q[2];
    unsigned short u[16];
};

// Native f32 -> bf16 (lowers to v_cvt_pk_bf16_f32 / v_cvt_bf16_f32 on gfx1250,
// replacing the 3-VALU manual round-to-nearest-even sequence).
__device__ __forceinline__ unsigned short f2bf(float f) {
    __bf16 b = (__bf16)f;
    return __builtin_bit_cast(unsigned short, b);
}

__device__ __forceinline__ float bf2f(unsigned short h) {
    return __uint_as_float((unsigned int)h << 16);
}

// Native transcendental tanh (v_tanh_f32) when available on this toolchain.
__device__ __forceinline__ float fast_tanh(float x) {
#if __has_builtin(__builtin_amdgcn_tanhf)
    return __builtin_amdgcn_tanhf(x);
#else
    return tanhf(x);
#endif
}

__device__ __forceinline__ v8f wmma_bf16(v16bf a, v16bf b, v8f c) {
    // (neg_a, A, neg_b, B, c_mod, C, reuse_a, reuse_b)
    return __builtin_amdgcn_wmma_f32_16x16x32_bf16(false, a, false, b,
                                                   (short)0, c, false, false);
}

// 16-bit fragment (A: 16x32 rows=M, or B: 32x16 rows=N of row-major W) from
// a row-major u16 LDS array.  Per ISA layout: lane half h, row = row0+lane%16,
// elems[0..7] = k0 + 8h .. +7, elems[8..15] = k0 + 16 + 8h .. +7.
__device__ __forceinline__ v16bf frag_lds(const unsigned short* base, int ldstr,
                                          int row0, int k0, int lane) {
    const int half = lane >> 4;
    const unsigned short* p = base + (row0 + (lane & 15)) * ldstr + k0 + half * 8;
    Frag f;
    f.q[0] = *(const uint4*)(p);
    f.q[1] = *(const uint4*)(p + 16);
    return f.v;
}

// Same fragment built from a row-major fp32 global array (converted to bf16).
__device__ __forceinline__ v16bf frag_gf32(const float* __restrict__ W, int ldw,
                                           int row0, int k0, int lane) {
    const int half = lane >> 4;
    const float* p = W + (size_t)(row0 + (lane & 15)) * ldw + k0 + half * 8;
    Frag f;
#pragma unroll
    for (int i = 0; i < 8; ++i) {
        f.u[i]     = f2bf(p[i]);
        f.u[8 + i] = f2bf(p[16 + i]);
    }
    return f.v;
}

__global__ __launch_bounds__(256)
void rnn_persistent_kernel(const int*   __restrict__ x,
                           const float* __restrict__ emb,
                           const float* __restrict__ W_hh,
                           const float* __restrict__ b_hh,
                           const float* __restrict__ W_xh,
                           const float* __restrict__ b_xh,
                           const float* __restrict__ W_hy,
                           const float* __restrict__ b_hy,
                           const float* __restrict__ h0,
                           float*       __restrict__ out) {
    extern __shared__ unsigned char smem[];
    unsigned short* P_lds    = (unsigned short*)(smem);                                  // 128 x HSTR bf16
    unsigned short* Why_lds  = (unsigned short*)(smem + (size_t)VOCAB * HSTR * 2);       // 128 x HSTR bf16
    unsigned short* h_lds    = (unsigned short*)(smem + (size_t)2 * VOCAB * HSTR * 2);   // 16 x HSTR bf16
    float*          base_lds = (float*)(smem + (size_t)2 * VOCAB * HSTR * 2 + 16 * HSTR * 2);
    int*            x_lds    = (int*)  (smem + (size_t)2 * VOCAB * HSTR * 2 + 16 * HSTR * 2 + 16 * BSTR * 4);

    const int tid     = threadIdx.x;
    const int lane    = tid & 31;
    const int w       = tid >> 5;        // wave id 0..7
    const int half    = lane >> 4;
    const int nlo     = lane & 15;
    const int rowbase = blockIdx.x * 16; // 16 batch rows per workgroup

    // ---- stage: x rows, W_hy (bf16), initial h (bf16) into LDS ----
    for (int i = tid; i < 16 * SEQ; i += 256) {
        int m = i >> 9, t = i & (SEQ - 1);
        x_lds[m * XSTR + t] = x[(size_t)(rowbase + m) * SEQ + t];
    }
    for (int i = tid; i < VOCAB * HIDDEN; i += 256) {
        int r = i >> 8, c = i & (HIDDEN - 1);
        Why_lds[r * HSTR + c] = f2bf(W_hy[i]);
    }
    for (int i = tid; i < 16 * HIDDEN; i += 256) {
        int m = i >> 8, c = i & (HIDDEN - 1);
        h_lds[m * HSTR + c] = f2bf(h0[c]);
    }

    // ---- P = emb @ W_xh^T via WMMA (128x256, K=128), bf16 into LDS ----
    for (int tile = w; tile < (VOCAB / 16) * (HIDDEN / 16); tile += 8) {
        const int mt = tile / (HIDDEN / 16);
        const int nt = tile % (HIDDEN / 16);
        v8f acc = {};
#pragma unroll
        for (int kt = 0; kt < EMBED / 32; ++kt) {
            v16bf a = frag_gf32(emb,  EMBED, mt * 16, kt * 32, lane);
            v16bf b = frag_gf32(W_xh, EMBED, nt * 16, kt * 32, lane);
            acc = wmma_bf16(a, b, acc);
        }
#pragma unroll
        for (int j = 0; j < 8; ++j) {
            int m = mt * 16 + j + 8 * half;
            P_lds[m * HSTR + nt * 16 + nlo] = f2bf(acc[j]);
        }
    }
    __syncthreads();

    // ---- baseline(m,c) = b_hh[c] + b_xh[c] + P[x[b,0]][c]  (constant over t) ----
    for (int i = tid; i < 16 * HIDDEN; i += 256) {
        int m = i >> 8, c = i & (HIDDEN - 1);
        int tok0 = x_lds[m * XSTR];
        base_lds[m * BSTR + c] = b_hh[c] + b_xh[c] + bf2f(P_lds[tok0 * HSTR + c]);
    }

    // ---- W_hh B-fragments resident in VGPRs for the whole time loop ----
    // wave w owns hidden column tiles 2w and 2w+1 (32 columns), 8 K-tiles each.
    v16bf Bhh[2][8];
#pragma unroll
    for (int ct = 0; ct < 2; ++ct)
#pragma unroll
        for (int kt = 0; kt < 8; ++kt)
            Bhh[ct][kt] = frag_gf32(W_hh, HIDDEN, (2 * w + ct) * 16, kt * 32, lane);

    const float by = b_hy[w * 16 + nlo];
    const int   c0 = (2 * w) * 16 + nlo;   // hidden col, tile 0
    const int   c1 = c0 + 16;              // hidden col, tile 1
    const int   vc = w * 16 + nlo;         // vocab col for y
    __syncthreads();

    // ================= sequential time loop (no inter-WG sync needed) =========
    for (int t = 0; t < NSTEP; ++t) {
        // --- h @ W_hh^T : 16 WMMAs per wave (2 col tiles x 8 K-tiles) ---
        v8f acc0 = {}, acc1 = {};
#pragma unroll
        for (int kt = 0; kt < 8; ++kt) {
            v16bf a = frag_lds(h_lds, HSTR, 0, kt * 32, lane);
            acc0 = wmma_bf16(a, Bhh[0][kt], acc0);
            acc1 = wmma_bf16(a, Bhh[1][kt], acc1);
        }

        // --- add input projection gathers + biases, tanh ---
        unsigned short hn0[8], hn1[8];
#pragma unroll
        for (int j = 0; j < 8; ++j) {
            int m   = j + 8 * half;          // D layout: M = j + 8*(lane>=16)
            int tok = x_lds[m * XSTR + t];
            float e0 = acc0[j] + base_lds[m * BSTR + c0] + bf2f(P_lds[tok * HSTR + c0]);
            float e1 = acc1[j] + base_lds[m * BSTR + c1] + bf2f(P_lds[tok * HSTR + c1]);
            hn0[j] = f2bf(fast_tanh(e0));
            hn1[j] = f2bf(fast_tanh(e1));
        }

        __syncthreads();   // everyone done reading h_old
#pragma unroll
        for (int j = 0; j < 8; ++j) {
            int m = j + 8 * half;
            h_lds[m * HSTR + c0] = hn0[j];
            h_lds[m * HSTR + c1] = hn1[j];
        }
        __syncthreads();   // h_new published

        // --- y = h_new @ W_hy^T : 8 WMMAs per wave (1 vocab tile) ---
        v8f accY = {};
#pragma unroll
        for (int kt = 0; kt < 8; ++kt) {
            v16bf a = frag_lds(h_lds,   HSTR, 0,      kt * 32, lane);
            v16bf b = frag_lds(Why_lds, HSTR, w * 16, kt * 32, lane);
            accY = wmma_bf16(a, b, accY);
        }
#pragma unroll
        for (int j = 0; j < 8; ++j) {
            int m = j + 8 * half;
            out[((size_t)(rowbase + m) * NSTEP + t) * VOCAB + vc] = accY[j] + by;
        }
    }
}

extern "C" void kernel_launch(void* const* d_in, const int* in_sizes, int n_in,
                              void* d_out, int out_size, void* d_ws, size_t ws_size,
                              hipStream_t stream) {
    (void)in_sizes; (void)n_in; (void)out_size; (void)d_ws; (void)ws_size;
    const int*   x    = (const int*)  d_in[0];
    const float* emb  = (const float*)d_in[1];
    const float* W_hh = (const float*)d_in[2];
    const float* b_hh = (const float*)d_in[3];
    const float* W_xh = (const float*)d_in[4];
    const float* b_xh = (const float*)d_in[5];
    const float* W_hy = (const float*)d_in[6];
    const float* b_hy = (const float*)d_in[7];
    const float* h0   = (const float*)d_in[8];
    float* out = (float*)d_out;

    const size_t shmem = (size_t)2 * VOCAB * HSTR * 2   // P + Why (bf16)
                       + (size_t)16 * HSTR * 2          // h (bf16)
                       + (size_t)16 * BSTR * 4          // baseline (f32)
                       + (size_t)16 * XSTR * 4;         // x rows (int)
    // 32 persistent workgroups x 256 threads (8 wave32), 16 batch rows each.
    rnn_persistent_kernel<<<dim3(BATCH / 16), dim3(256), shmem, stream>>>(
        x, emb, W_hh, b_hh, W_xh, b_xh, W_hy, b_hy, h0, out);
}
